// MultiheadAttentionCustom_89919435309483
// MI455X (gfx1250) — compile-verified, matrix operand order checked by reading, not run on hardware
//
#include <hip/hip_runtime.h>

// Multihead attention with relative position bias for MI455X (gfx1250).
// All GEMMs via v_wmma_f32_16x16x32_f16 (f16 in, f32 accumulate).
// fp32 operands are converted to f16 ONCE (bandwidth-bound pass), so the
// GEMM inner loops are pure global_load_b128/ds_load_b128 + v_wmma.
// L = S = 2048, N = 2, E = 1024, H = 16, D = 64.

#define L_DIM   2048
#define S_DIM   2048
#define N_BATCH 2
#define E_DIM   1024
#define H_NUM   16
#define D_HEAD  64
#define BIAS_W  (2 * S_DIM - 1)   // 4095

typedef __attribute__((ext_vector_type(16))) _Float16 v16h;
typedef __attribute__((ext_vector_type(8)))  _Float16 v8h;
typedef __attribute__((ext_vector_type(8)))  float    v8f;

__device__ __forceinline__ v8f wmma_f16(v16h a, v16h b, v8f c) {
  // D = A(16x32 f16) * B(32x16 f16) + C(16x16 f32)
  return __builtin_amdgcn_wmma_f32_16x16x32_f16(
      /*neg_a=*/false, a, /*neg_b=*/false, b,
      /*c_mod=*/(short)0, c, /*reuse_a=*/false, /*reuse_b=*/false);
}

// Load a 16(row) x 32(K) f16 fragment whose rows are contiguous-K in memory.
// Per ISA 16-bit A layout: lane (row = lane&15, half = lane>>4) holds
// K = [8*half .. +7] and [16+8*half .. +7] (two 16B runs). The identical
// layout serves as a B fragment with "row" = N column.  Works for global
// or LDS pointers (LDS -> ds_load_b128 after address-space inference).
__device__ __forceinline__ v16h frag_f16(const _Float16* __restrict__ p, int ld) {
  const int lane = threadIdx.x & 31;
  const _Float16* r = p + (lane & 15) * ld + (lane >> 4) * 8;
  v8h lo = *reinterpret_cast<const v8h*>(r);
  v8h hi = *reinterpret_cast<const v8h*>(r + 16);
  v16h f;
#pragma unroll
  for (int i = 0; i < 8; ++i) { f[i] = lo[i]; f[8 + i] = hi[i]; }
  return f;
}

// ---------------------------------------------------------------------------
// Kernel 0: one-shot fp32 -> f16 conversion (8 elements / thread, b128 I/O).
// ---------------------------------------------------------------------------
__global__ __launch_bounds__(256) void cvt_f16_kernel(
    const float* __restrict__ src, _Float16* __restrict__ dst, int n8) {
  const int i = blockIdx.x * 256 + threadIdx.x;
  if (i >= n8) return;
  float4 a = reinterpret_cast<const float4*>(src)[2 * i];
  float4 b = reinterpret_cast<const float4*>(src)[2 * i + 1];
  v8h o;
  o[0] = (_Float16)a.x; o[1] = (_Float16)a.y; o[2] = (_Float16)a.z; o[3] = (_Float16)a.w;
  o[4] = (_Float16)b.x; o[5] = (_Float16)b.y; o[6] = (_Float16)b.z; o[7] = (_Float16)b.w;
  reinterpret_cast<v8h*>(dst)[i] = o;
}

// ---------------------------------------------------------------------------
// Kernel 1: fused QKV projection.  grid.z: 0=Q, 1=K, 2=V.  All-f16 operands.
// Y[t, c] = X[t, :] . W[z*E + c, :] + bias[z*E + c]   (t = l*N + n)
// Q scaled by D^-0.5 -> qh[n][h][l][d]; K -> kh[n][h][l][d];
// V -> TRANSPOSED vt[n][h][d][s].
// Block 256 thr (8 waves) -> 128x128 tile; wave = 32x64 (2x4 WMMA tiles).
// ---------------------------------------------------------------------------
__global__ __launch_bounds__(256) void qkv_proj_kernel(
    const _Float16* __restrict__ xq, const _Float16* __restrict__ xk,
    const _Float16* __restrict__ xv, const _Float16* __restrict__ Wf,
    const float* __restrict__ bias,
    _Float16* __restrict__ qh, _Float16* __restrict__ kh,
    _Float16* __restrict__ vt) {
  const int z = blockIdx.z;
  const _Float16* X = (z == 0) ? xq : (z == 1) ? xk : xv;

  const int w  = threadIdx.x >> 5;
  const int wm = w >> 1, wn = w & 1;
  const int m0 = blockIdx.x * 128 + wm * 32;      // token rows
  const int n0 = blockIdx.y * 128 + wn * 64;      // output cols (within E)
  const _Float16* Wp = Wf + ((size_t)(z * E_DIM + n0)) * E_DIM;
  const _Float16* Ap = X + (size_t)m0 * E_DIM;
  const float* bz = bias + z * E_DIM;

  v8f acc[2][4] = {};
  for (int k0 = 0; k0 < E_DIM; k0 += 32) {
    if (k0 + 32 < E_DIM) {
      __builtin_prefetch(Ap + k0 + 32, 0, 0);
      __builtin_prefetch(Wp + k0 + 32, 0, 0);
    }
    v16h a0 = frag_f16(Ap + k0, E_DIM);
    v16h a1 = frag_f16(Ap + (size_t)16 * E_DIM + k0, E_DIM);
#pragma unroll
    for (int j = 0; j < 4; ++j) {
      v16h b = frag_f16(Wp + (size_t)(j * 16) * E_DIM + k0, E_DIM);
      acc[0][j] = wmma_f16(a0, b, acc[0][j]);
      acc[1][j] = wmma_f16(a1, b, acc[1][j]);
    }
  }

  // Epilogue: C layout -> row = g + 8*half (VGPR index), col = lane&15.
  const int lane = threadIdx.x & 31;
  const int col  = lane & 15;
  const int half = lane >> 4;
#pragma unroll
  for (int i = 0; i < 2; ++i) {
#pragma unroll
    for (int j = 0; j < 4; ++j) {
#pragma unroll
      for (int g = 0; g < 8; ++g) {
        const int t = m0 + i * 16 + g + half * 8;   // token index = l*N + n
        const int c = n0 + j * 16 + col;            // column within E
        float v = acc[i][j][g] + bz[c];
        const int l = t >> 1, n = t & 1;            // N_BATCH == 2
        const int h = c >> 6, d = c & 63;           // D_HEAD == 64
        const int nh = n * H_NUM + h;
        if (z == 0)
          qh[((size_t)nh * L_DIM + l) * D_HEAD + d] = (_Float16)(v * 0.125f);
        else if (z == 1)
          kh[((size_t)nh * S_DIM + l) * D_HEAD + d] = (_Float16)v;
        else
          vt[((size_t)nh * D_HEAD + d) * S_DIM + l] = (_Float16)v;
      }
    }
  }
}

// ---------------------------------------------------------------------------
// Kernel 2: flash attention, one (n,h) per blockIdx.y, 128 query rows/block.
// K/V tiles staged in LDS once per block (8x traffic reduction vs per-wave).
// Each wave owns 16 query rows.  Computes S^T = K . Q^T so that:
//   - softmax stats are per-lane scalars (lane <-> query column),
//   - exp(S^T) tiles are bit-exact the fragment layout required for
//     O^T = V^T . P^T  (no transpose / shuffle needed).
// ---------------------------------------------------------------------------
__global__ __launch_bounds__(256) void flash_attn_kernel(
    const _Float16* __restrict__ qh, const _Float16* __restrict__ kh,
    const _Float16* __restrict__ vt, const float* __restrict__ rel_bias,
    _Float16* __restrict__ ch) {
  __shared__ __align__(16) _Float16 lk[32 * D_HEAD];   // K tile: 32 s x 64 d
  __shared__ __align__(16) _Float16 lv[D_HEAD * 32];   // V^T tile: 64 d x 32 s

  const int nh   = blockIdx.y;            // n*H + h
  const int h    = nh & (H_NUM - 1);
  const int n    = nh >> 4;
  const int w    = threadIdx.x >> 5;
  const int lane = threadIdx.x & 31;
  const int q    = lane & 15;             // query column owned by this lane
  const int half = lane >> 4;
  const int qbase = blockIdx.x * 128 + w * 16;
  const int lglob = qbase + q;            // global query row

  const _Float16* Qp = qh + ((size_t)nh * L_DIM + qbase) * D_HEAD;
  const _Float16* Kp = kh + (size_t)nh * S_DIM * D_HEAD;
  const _Float16* Vp = vt + (size_t)nh * D_HEAD * S_DIM;
  // bias[h, l - s + S-1] accessed as bp[-s]
  const float* bp = rel_bias + (size_t)h * BIAS_W + lglob + (S_DIM - 1);

  // Cooperative-load coordinates (256 threads x 16B covers each 4KB tile).
  const int tid  = threadIdx.x;
  const int krow = tid >> 3, kcol = (tid & 7) << 3;   // 32 x 64 halfs
  const int vrow = tid >> 2, vcol = (tid & 3) << 3;   // 64 x 32 halfs

  // Q^T B-fragments (lane row = query, contiguous d), loaded once.
  const v16h qb0 = frag_f16(Qp, D_HEAD);
  const v16h qb1 = frag_f16(Qp + 32, D_HEAD);

  v8f ot0 = {}, ot1 = {}, ot2 = {}, ot3 = {};   // O^T: VGPR<->d, lane<->query
  float mrow = -3.0e38f, srow = 0.0f;

  for (int s0 = 0; s0 < S_DIM; s0 += 32) {
    // Stage K/V tiles: global -> regs (overlaps prior compute) -> LDS.
    v8h kreg = *reinterpret_cast<const v8h*>(
        Kp + (size_t)(s0 + krow) * D_HEAD + kcol);
    v8h vreg = *reinterpret_cast<const v8h*>(
        Vp + (size_t)vrow * S_DIM + s0 + vcol);
    if (s0 + 32 < S_DIM) {
      __builtin_prefetch(Kp + (size_t)(s0 + 32 + krow) * D_HEAD + kcol, 0, 0);
      __builtin_prefetch(Vp + (size_t)vrow * S_DIM + s0 + 32 + vcol, 0, 0);
    }
    __syncthreads();   // previous tile fully consumed
    *reinterpret_cast<v8h*>(lk + krow * D_HEAD + kcol) = kreg;
    *reinterpret_cast<v8h*>(lv + vrow * 32 + vcol) = vreg;
    __syncthreads();   // tile visible to all waves

    // S^T = K . Q^T  (A rows = key s, contiguous d from LDS).
    v16h ka0 = frag_f16(lk, D_HEAD);
    v16h ka1 = frag_f16(lk + 32, D_HEAD);
    v16h ka2 = frag_f16(lk + 16 * D_HEAD, D_HEAD);
    v16h ka3 = frag_f16(lk + 16 * D_HEAD + 32, D_HEAD);

    v8f st0 = {}, st1 = {};
    st0 = wmma_f16(ka0, qb0, st0);
    st0 = wmma_f16(ka1, qb1, st0);   // keys s0..s0+15
    st1 = wmma_f16(ka2, qb0, st1);
    st1 = wmma_f16(ka3, qb1, st1);   // keys s0+16..s0+31

    // Relative position bias; key index s = s0 + g + 8*half (+16 for st1).
    float sv0[8], sv1[8];
#pragma unroll
    for (int g = 0; g < 8; ++g) {
      const int so = g + half * 8;
      sv0[g] = st0[g] + bp[-(s0 + so)];
      sv1[g] = st1[g] + bp[-(s0 + so + 16)];
    }

    // Online softmax (per-query stats per-lane; merge the two lane-halves).
    float tmax = sv0[0];
#pragma unroll
    for (int g = 0; g < 8; ++g) {
      tmax = fmaxf(tmax, sv0[g]);
      tmax = fmaxf(tmax, sv1[g]);
    }
    tmax = fmaxf(tmax, __shfl_xor(tmax, 16));
    const float mnew  = fmaxf(mrow, tmax);
    const float alpha = __expf(mrow - mnew);
    mrow = mnew;

    // P^T fragment: element e<8 -> k=8*half+e (st0[e]); e>=8 -> st1[e-8].
    v16h pa;
    float tsum = 0.0f;
#pragma unroll
    for (int g = 0; g < 8; ++g) {
      const float p0 = __expf(sv0[g] - mnew);
      const float p1 = __expf(sv1[g] - mnew);
      tsum += p0 + p1;
      pa[g]     = (_Float16)p0;
      pa[8 + g] = (_Float16)p1;
    }
    tsum += __shfl_xor(tsum, 16);
    srow = srow * alpha + tsum;

#pragma unroll
    for (int g = 0; g < 8; ++g) {
      ot0[g] *= alpha; ot1[g] *= alpha; ot2[g] *= alpha; ot3[g] *= alpha;
    }

    // O^T += V^T . P^T   (A rows = d, contiguous s from LDS; B = pa)
    ot0 = wmma_f16(frag_f16(lv + 0 * 16 * 32, 32), pa, ot0);
    ot1 = wmma_f16(frag_f16(lv + 1 * 16 * 32, 32), pa, ot1);
    ot2 = wmma_f16(frag_f16(lv + 2 * 16 * 32, 32), pa, ot2);
    ot3 = wmma_f16(frag_f16(lv + 3 * 16 * 32, 32), pa, ot3);
  }

  // Normalize and scatter to context ch[t][E]; d = j*16 + half*8 + g is
  // contiguous in g -> one 16B store per tile.
  const float inv = 1.0f / srow;
  const int t = lglob * N_BATCH + n;
  _Float16* op = ch + (size_t)t * E_DIM + h * D_HEAD + half * 8;
  v8f* ots[4] = {&ot0, &ot1, &ot2, &ot3};
#pragma unroll
  for (int j = 0; j < 4; ++j) {
    v8h o;
#pragma unroll
    for (int g = 0; g < 8; ++g) o[g] = (_Float16)((*ots[j])[g] * inv);
    *reinterpret_cast<v8h*>(op + j * 16) = o;
  }
}

// ---------------------------------------------------------------------------
// Kernel 3: output projection.  out[t, e] = ch[t, :] . Wout[e, :] + bout[e]
// Same 128x128 block / 32x64 wave tiling as the QKV projection.
// ---------------------------------------------------------------------------
__global__ __launch_bounds__(256) void out_proj_kernel(
    const _Float16* __restrict__ ch, const _Float16* __restrict__ Wof,
    const float* __restrict__ bout, float* __restrict__ out) {
  const int w  = threadIdx.x >> 5;
  const int wm = w >> 1, wn = w & 1;
  const int m0 = blockIdx.x * 128 + wm * 32;
  const int n0 = blockIdx.y * 128 + wn * 64;
  const _Float16* Wp = Wof + (size_t)n0 * E_DIM;
  const _Float16* Ap = ch + (size_t)m0 * E_DIM;

  v8f acc[2][4] = {};
  for (int k0 = 0; k0 < E_DIM; k0 += 32) {
    if (k0 + 32 < E_DIM) {
      __builtin_prefetch(Ap + k0 + 32, 0, 0);
      __builtin_prefetch(Wp + k0 + 32, 0, 0);
    }
    v16h a0 = frag_f16(Ap + k0, E_DIM);
    v16h a1 = frag_f16(Ap + (size_t)16 * E_DIM + k0, E_DIM);
#pragma unroll
    for (int j = 0; j < 4; ++j) {
      v16h b = frag_f16(Wp + (size_t)(j * 16) * E_DIM + k0, E_DIM);
      acc[0][j] = wmma_f16(a0, b, acc[0][j]);
      acc[1][j] = wmma_f16(a1, b, acc[1][j]);
    }
  }

  const int lane = threadIdx.x & 31;
  const int col  = lane & 15;
  const int half = lane >> 4;
#pragma unroll
  for (int i = 0; i < 2; ++i) {
#pragma unroll
    for (int j = 0; j < 4; ++j) {
#pragma unroll
      for (int g = 0; g < 8; ++g) {
        const int t = m0 + i * 16 + g + half * 8;
        const int c = n0 + j * 16 + col;
        out[(size_t)t * E_DIM + c] = acc[i][j][g] + bout[c];
      }
    }
  }
}

// ---------------------------------------------------------------------------
extern "C" void kernel_launch(void* const* d_in, const int* in_sizes, int n_in,
                              void* d_out, int out_size, void* d_ws, size_t ws_size,
                              hipStream_t stream) {
  const float* query = (const float*)d_in[0];
  const float* key   = (const float*)d_in[1];
  const float* value = (const float*)d_in[2];
  const float* ipw   = (const float*)d_in[3];
  const float* ipb   = (const float*)d_in[4];
  const float* opw   = (const float*)d_in[5];
  const float* opb   = (const float*)d_in[6];
  const float* rbias = (const float*)d_in[7];
  float* out = (float*)d_out;

  // Workspace layout (f16 elements):
  //   qh | kh | vt | ch : N*H*L*D = 4 Mi each
  //   xq | xk | xv      : L*N*E   = 4 Mi each      (f16 activations)
  //   wf                : 3*E*E   = 3 Mi           (f16 in_proj_weight)
  //   wof               : E*E     = 1 Mi           (f16 out_proj_weight)
  // Total 32 Mi f16 = 64 MiB.
  const size_t slab = (size_t)N_BATCH * H_NUM * L_DIM * D_HEAD;  // 4 Mi
  _Float16* qh  = (_Float16*)d_ws;
  _Float16* kh  = qh + slab;
  _Float16* vt  = kh + slab;
  _Float16* ch  = vt + slab;
  _Float16* xq  = ch + slab;
  _Float16* xk  = xq + slab;
  _Float16* xv  = xk + slab;
  _Float16* wf  = xv + slab;
  _Float16* wof = wf + (size_t)3 * E_DIM * E_DIM;

  dim3 blk(256);
  const int act8 = (int)(slab / 8);                    // 524288
  const int ipw8 = 3 * E_DIM * E_DIM / 8;              // 393216
  const int opw8 = E_DIM * E_DIM / 8;                  // 131072
  cvt_f16_kernel<<<dim3(act8 / 256), blk, 0, stream>>>(query, xq, act8);
  cvt_f16_kernel<<<dim3(act8 / 256), blk, 0, stream>>>(key,   xk, act8);
  cvt_f16_kernel<<<dim3(act8 / 256), blk, 0, stream>>>(value, xv, act8);
  cvt_f16_kernel<<<dim3(ipw8 / 256), blk, 0, stream>>>(ipw,   wf, ipw8);
  cvt_f16_kernel<<<dim3(opw8 / 256), blk, 0, stream>>>(opw,   wof, opw8);

  // QKV projection: M=4096 -> 32 tiles, Nc=1024 -> 8 tiles, z = {Q,K,V}.
  qkv_proj_kernel<<<dim3(32, 8, 3), blk, 0, stream>>>(
      xq, xk, xv, wf, ipb, qh, kh, vt);
  // Flash attention: 16 query blocks x 32 (n,h) slices.
  flash_attn_kernel<<<dim3(L_DIM / 128, N_BATCH * H_NUM), blk, 0, stream>>>(
      qh, kh, vt, rbias, ch);
  // Output projection.
  out_proj_kernel<<<dim3(32, 8), blk, 0, stream>>>(ch, opw ? wof : wof, opb, out);
}
